// Self_Attn_36928128811059
// MI455X (gfx1250) — matrix operand order repaired and checked by reading, not compiled, not run on hardware
//
#include <hip/hip_runtime.h>

#define IN_DIM 128
#define KEY_F  64
#define VAL_F  128
#define BATCH  8
#define SEQ    2048

#define LOG2E 1.4426950408889634f

typedef __attribute__((ext_vector_type(16))) __bf16 v16bf;
typedef __attribute__((ext_vector_type(8)))  float  v8f;

// A-fragment block permutation: within each 32-feature chunk, reorder the
// 8-wide blocks [b0 b1 b2 b3] -> [b0 b2 b1 b3] so that one lane's 16
// fragment elements {8h..8h+7, 16+8h..23+8h} are contiguous in memory.
__device__ __forceinline__ int afrag_pos(int fo) {
  int bi = (fo >> 3) & 3;
  int pb = ((bi & 1) << 1) | (bi >> 1);      // swap the two block-index bits
  return (fo & ~31) | (pb << 3) | (fo & 7);
}

// -------------------------------------------------------------------------
// Projection kernel: y = W x + b for q/k/v, output bf16:
//   qT   [B][L][64]   natural f order, pre-scaled by log2e -> QK A-frags
//   kSwz [B][4][L][16] ((f>>4), l, f&15)                   -> QK B-frags
//   vNat [B][128][L]  (f major)                            -> PV B-frags
// -------------------------------------------------------------------------
__global__ __launch_bounds__(256) void proj_kernel(
    const float* __restrict__ x,
    const float* __restrict__ Wq, const float* __restrict__ bq,
    const float* __restrict__ Wk, const float* __restrict__ bk,
    const float* __restrict__ Wv, const float* __restrict__ bv,
    __bf16* __restrict__ qT, __bf16* __restrict__ kSwz,
    __bf16* __restrict__ vNat)
{
  __shared__ float xs[IN_DIM * 64];
  const int b   = blockIdx.y;
  const int l0  = blockIdx.x * 64;
  const int tid = threadIdx.x;

  for (int idx = tid; idx < IN_DIM * 64; idx += 256) {
    int c = idx >> 6, lq = idx & 63;
    xs[idx] = x[((size_t)b * IN_DIM + c) * SEQ + l0 + lq];
  }
  __syncthreads();

  const int lq   = tid & 63;
  const int fgrp = tid >> 6;           // 0..3 (wave-uniform)
  const int l    = l0 + lq;

  for (int ff = 0; ff < 64; ++ff) {
    int f = fgrp * 64 + ff;            // 0..255: [0,64)=q [64,128)=k [128,256)=v
    const float* W;
    float bias;
    int fo;
    if (f < 64)       { W = Wq; fo = f;       bias = bq[fo]; }
    else if (f < 128) { W = Wk; fo = f - 64;  bias = bk[fo]; }
    else              { W = Wv; fo = f - 128; bias = bv[fo]; }
    float acc = bias;
#pragma unroll 8
    for (int c = 0; c < IN_DIM; ++c)
      acc += W[fo * IN_DIM + c] * xs[c * 64 + lq];
    if (f < 64)        // fold log2e into Q so softmax can use raw exp2
      qT[((size_t)b * SEQ + l) * KEY_F + fo] = (__bf16)(acc * LOG2E);
    else if (f < 128)
      kSwz[(((size_t)b * 4 + (fo >> 4)) * SEQ + l) * 16 + (fo & 15)] = (__bf16)acc;
    else
      vNat[((size_t)b * VAL_F + fo) * SEQ + l] = (__bf16)acc;
  }
}

// -------------------------------------------------------------------------
// Flash attention: each wave owns a 16-row query tile, streams keys/values
// in 128-wide steps with online softmax (base-2). 52 v_wmma per step; row
// sums accumulate via WMMA against an all-ones B matrix. All hot-loop
// memory ops are [per-lane base + constant immediate].
// -------------------------------------------------------------------------
__global__ __launch_bounds__(128) void attn_kernel(
    const __bf16* __restrict__ qT, const __bf16* __restrict__ kSwz,
    const __bf16* __restrict__ vNat, const float* __restrict__ x,
    const float* __restrict__ gammap, float* __restrict__ y)
{
  __shared__ __bf16 pLds[4][16 * 128];   // 4 KB per wave P staging tile

  const int b    = blockIdx.y;
  const int wave = threadIdx.x >> 5;
  const int lane = threadIdx.x & 31;
  const int h    = lane >> 4;           // half-wave select
  const int n    = lane & 15;           // col (B/C frags) or row (A frag)
  const int q0   = blockIdx.x * 64 + wave * 16;

  // ---- Q A-fragments: contiguous 16-feature slices of natural-order qT ----
  const __bf16* qrow = qT + ((size_t)b * SEQ + (q0 + n)) * KEY_F;
  v16bf aQ0 = *(const v16bf*)(qrow + 16 * h);
  v16bf aQ1 = *(const v16bf*)(qrow + 32 + 16 * h);

  // all-ones B fragment for row-sum accumulation via WMMA
  v16bf bOne;
#pragma unroll
  for (int e = 0; e < 16; ++e) bOne[e] = (__bf16)1.0f;

  v8f O[8] = {};                         // 16x128 f32 output accumulator
  v8f Ol = {};                           // row-sum accumulator (replicated cols)
  float m[8];
#pragma unroll
  for (int g = 0; g < 8; ++g) m[g] = -1e30f;

  // ---- per-lane byte base pointers (hot loop uses constant offsets) ----
  const char* kp0 = (const char*)kSwz
      + (((size_t)b * 4 + h)       * SEQ + n) * 32;          // fg = h
  const char* kp1 = (const char*)kSwz
      + (((size_t)b * 4 + 2 + h)   * SEQ + n) * 32;          // fg = 2+h
  const char* vp  = (const char*)vNat
      + ((size_t)b * VAL_F + n) * (SEQ * 2);                 // f = 16t + n
  char* pwS0 = (char*)&pLds[wave][8 * h * 128] + 2 * 0;      // colOff applied below
  char* pwS1;
  {
    // per-lane column offsets for the natural-order P staging store
    int c0 = n;
    int c1 = 16 + n;
    pwS0 = (char*)&pLds[wave][8 * h * 128 + c0];
    pwS1 = (char*)&pLds[wave][8 * h * 128 + c1];
  }
  const char* pwA = (const char*)&pLds[wave][n * 128 + 16 * h];

  for (int j0 = 0; j0 < SEQ; j0 += 128) {
    const int jb32 = j0 * 32;            // byte offset into kSwz stream
    const int jb2  = j0 * 2;             // byte offset into vNat stream

    // ---- S = Q^T K : eight 16x16 tiles (cols j0+16*tj), K=64 ----
    v8f c[8];
#pragma unroll
    for (int tj = 0; tj < 8; ++tj) {
      v16bf bK0 = *(const v16bf*)(kp0 + jb32 + tj * 512);
      v16bf bK1 = *(const v16bf*)(kp1 + jb32 + tj * 512);
      v8f s0 = {};
      s0 = __builtin_amdgcn_wmma_f32_16x16x32_bf16(false, aQ0, false, bK0, (short)0, s0, false, false);
      s0 = __builtin_amdgcn_wmma_f32_16x16x32_bf16(false, aQ1, false, bK1, (short)0, s0, false, false);
      c[tj] = s0;
    }

    // ---- online softmax over the 128 new keys (row = g + 8h, base-2) ----
#pragma unroll
    for (int g = 0; g < 8; ++g) {
      float rm = fmaxf(fmaxf(fmaxf(c[0][g], c[1][g]), fmaxf(c[2][g], c[3][g])),
                       fmaxf(fmaxf(c[4][g], c[5][g]), fmaxf(c[6][g], c[7][g])));
      rm = fmaxf(rm, __shfl_xor(rm, 1));
      rm = fmaxf(rm, __shfl_xor(rm, 2));
      rm = fmaxf(rm, __shfl_xor(rm, 4));
      rm = fmaxf(rm, __shfl_xor(rm, 8));
      float mn = fmaxf(m[g], rm);
      float s  = __builtin_amdgcn_exp2f(m[g] - mn);
      m[g] = mn;
#pragma unroll
      for (int tj = 0; tj < 8; ++tj)
        c[tj][g] = __builtin_amdgcn_exp2f(c[tj][g] - mn);
      Ol[g] *= s;
#pragma unroll
      for (int t = 0; t < 8; ++t) O[t][g] *= s;
    }

    // ---- P (f32 C-layout) -> LDS bf16 (natural-order 16x128 rows) ----
#pragma unroll
    for (int tj = 0; tj < 8; ++tj) {
      char* base = (tj & 1) ? pwS1 : pwS0;
#pragma unroll
      for (int g = 0; g < 8; ++g)
        *(__bf16*)(base + (g * 128 + (tj >> 1) * 32) * 2) = (__bf16)c[tj][g];
    }
    asm volatile("s_wait_dscnt 0" ::: "memory");

    // ---- A-fragments of P: 4 contiguous 32B LDS loads per lane ----
    v16bf aP[4];
#pragma unroll
    for (int kc = 0; kc < 4; ++kc)
      aP[kc] = *(const v16bf*)(pwA + kc * 64);

    // ---- row sums: Ol += P * ones (replicated across columns) ----
#pragma unroll
    for (int kc = 0; kc < 4; ++kc)
      Ol = __builtin_amdgcn_wmma_f32_16x16x32_bf16(false, aP[kc], false, bOne, (short)0, Ol, false, false);

    // ---- O += P * V^T : 8 N-tiles over f=0..127, K=128 (4 chunks) ----
#pragma unroll
    for (int t = 0; t < 8; ++t) {
#pragma unroll
      for (int kc = 0; kc < 4; ++kc) {
        v16bf bV = *(const v16bf*)(vp + t * (16 * SEQ * 2) + jb2 + kc * 64 + h * 32);
        O[t] = __builtin_amdgcn_wmma_f32_16x16x32_bf16(false, aP[kc], false, bV, (short)0, O[t], false, false);
      }
    }
  }

  // ---- epilogue: y[b][f][i] = gamma * O[i][f]/l[i] + x[b][f][i] ----
  const float gamma = gammap[0];
  float inv_l[8];
#pragma unroll
  for (int g = 0; g < 8; ++g) inv_l[g] = 1.0f / Ol[g];

  const char* xb = (const char*)x
      + (((size_t)b * VAL_F + n) * SEQ + q0 + 8 * h) * 4;
  char* yb = (char*)y
      + (((size_t)b * VAL_F + n) * SEQ + q0 + 8 * h) * 4;
#pragma unroll
  for (int t = 0; t < 8; ++t) {
#pragma unroll
    for (int g = 0; g < 8; ++g) {
      int off = t * (16 * SEQ * 4) + g * 4;
      float xv = *(const float*)(xb + off);
      *(float*)(yb + off) = gamma * (O[t][g] * inv_l[g]) + xv;
    }
  }
}

// -------------------------------------------------------------------------
extern "C" void kernel_launch(void* const* d_in, const int* in_sizes, int n_in,
                              void* d_out, int out_size, void* d_ws, size_t ws_size,
                              hipStream_t stream) {
  const float* x     = (const float*)d_in[0];
  const float* Wq    = (const float*)d_in[1];
  const float* bq    = (const float*)d_in[2];
  const float* Wk    = (const float*)d_in[3];
  const float* bk    = (const float*)d_in[4];
  const float* Wv    = (const float*)d_in[5];
  const float* bv    = (const float*)d_in[6];
  const float* gamma = (const float*)d_in[7];
  float* y = (float*)d_out;

  char* ws = (char*)d_ws;
  const size_t qBytes = (size_t)BATCH * SEQ * KEY_F * sizeof(__bf16);   // 2 MiB
  __bf16* qT   = (__bf16*)ws;
  __bf16* kSwz = (__bf16*)(ws + qBytes);
  __bf16* vNat = (__bf16*)(ws + 2 * qBytes);                            // 8 MiB

  dim3 grid(SEQ / 64, BATCH);
  proj_kernel<<<grid, 256, 0, stream>>>(x, Wq, bq, Wk, bk, Wv, bv, qT, kSwz, vNat);
  attn_kernel<<<grid, 128, 0, stream>>>(qT, kSwz, vNat, x, gamma, y);
}
